// AttentionGate_11647951307116
// MI455X (gfx1250) — compile-verified, matrix operand order checked by reading, not run on hardware
//
#include <hip/hip_runtime.h>

// ---------------------------------------------------------------------------
// Types for CDNA5 WMMA (gfx1250, wave32)
// ---------------------------------------------------------------------------
typedef __bf16 bf16;
typedef __attribute__((ext_vector_type(16))) __bf16 v16bf;
typedef __attribute__((ext_vector_type(8)))  float  v8f;
typedef __attribute__((ext_vector_type(4)))  int    v4i;

// Pointer types matching the async-to-LDS builtin's parameters
// (param0: int4 addrspace(1)*, param1: int4 addrspace(3)* per hipcc diagnostic)
typedef __attribute__((address_space(1))) v4i* gv4i_p;
typedef __attribute__((address_space(3))) v4i* lv4i_p;

#define WMMA_BF16(a, b, c) \
    __builtin_amdgcn_wmma_f32_16x16x32_bf16(false, (a), false, (b), (short)0, (c), false, false)

#if __has_builtin(__builtin_amdgcn_global_load_async_to_lds_b128) && \
    __has_builtin(__builtin_amdgcn_s_wait_asynccnt)
#define USE_ASYNC_LDS 1
#else
#define USE_ASYNC_LDS 0
#endif

__device__ __forceinline__ v16bf ldfrag(const bf16* p) {
    // 32-byte aligned LDS fragment load (16 bf16)
    return *(const v16bf*)p;
}

// Problem constants
static constexpr int D  = 512;       // model dim (= INPUT_SIZE = 2*HIDDEN)
static constexpr int S  = 2048;      // sequence length
static constexpr int NB = 8;         // batch
static constexpr int M_TOT = NB * S; // 16384 rows for the projections

// ---------------------------------------------------------------------------
// Kernel 1: projection GEMM  C[m,n] = sum_k A[m,k] * W[k,n] + bias[n]
// A: f32 [M,512], W: f32 [512,512] row-major (k-major), C: bf16 [M,512]
// Tile: 128(M) x 128(N), KT=32.  256 threads = 8 waves; wave -> 64x32 subtile.
// ---------------------------------------------------------------------------
static constexpr int P_STRIDE = 48;  // padded LDS row stride; 96B => 32B-aligned frags

__global__ __launch_bounds__(256)
void proj_gemm_bf16(const float* __restrict__ A, const float* __restrict__ W,
                    const float* __restrict__ bias, bf16* __restrict__ C) {
    __shared__ bf16 As[128 * P_STRIDE];   // [row m 0..127][k 0..31]
    __shared__ bf16 Ws[128 * P_STRIDE];   // transposed: [col n 0..127][k 0..31]

    const int n0 = blockIdx.x * 128;
    const int m0 = blockIdx.y * 128;

    const int tid  = threadIdx.x;
    const int lane = tid & 31;
    const int wave = tid >> 5;
    const int ln   = lane & 15;        // N / M index within fragment
    const int hi   = lane >> 4;        // K-half selector (A/B frag layout)
    const int wm   = wave >> 2;        // 0..1 : 64-row slab
    const int wn   = wave & 3;         // 0..3 : 32-col slab

    v8f acc[4][2];
    #pragma unroll
    for (int mt = 0; mt < 4; ++mt)
        #pragma unroll
        for (int nt = 0; nt < 2; ++nt) acc[mt][nt] = {};

    for (int kk = 0; kk < 16; ++kk) {       // K = 512, KT = 32
        const int k0 = kk * 32;
        // ---- stage A tile: 128x32 f32 -> bf16, row-major ----
        #pragma unroll
        for (int p = 0; p < 4; ++p) {
            const int r = (tid >> 3) + p * 32;       // 0..127
            const int c = (tid & 7) * 4;             // 0..28
            float4 f = *(const float4*)&A[(size_t)(m0 + r) * D + k0 + c];
            bf16* dst = &As[r * P_STRIDE + c];
            dst[0] = (bf16)f.x; dst[1] = (bf16)f.y;
            dst[2] = (bf16)f.z; dst[3] = (bf16)f.w;
        }
        // ---- stage W tile transposed: W[k, n] -> Ws[n][k] ----
        #pragma unroll
        for (int p = 0; p < 4; ++p) {
            const int r = (tid >> 5) + p * 8;        // k row 0..31
            const int c = (tid & 31) * 4;            // n col 0..124
            float4 f = *(const float4*)&W[(size_t)(k0 + r) * D + n0 + c];
            Ws[(c + 0) * P_STRIDE + r] = (bf16)f.x;
            Ws[(c + 1) * P_STRIDE + r] = (bf16)f.y;
            Ws[(c + 2) * P_STRIDE + r] = (bf16)f.z;
            Ws[(c + 3) * P_STRIDE + r] = (bf16)f.w;
        }
        __syncthreads();

        // ---- fragments + 8 WMMAs ----
        v16bf af[4], bf_[2];
        #pragma unroll
        for (int mt = 0; mt < 4; ++mt)
            af[mt] = ldfrag(&As[(wm * 64 + mt * 16 + ln) * P_STRIDE + hi * 16]);
        #pragma unroll
        for (int nt = 0; nt < 2; ++nt)
            bf_[nt] = ldfrag(&Ws[(wn * 32 + nt * 16 + ln) * P_STRIDE + hi * 16]);
        #pragma unroll
        for (int mt = 0; mt < 4; ++mt)
            #pragma unroll
            for (int nt = 0; nt < 2; ++nt)
                acc[mt][nt] = WMMA_BF16(af[mt], bf_[nt], acc[mt][nt]);
        __syncthreads();
    }

    // ---- epilogue: bias add, convert to bf16, store ----
    #pragma unroll
    for (int mt = 0; mt < 4; ++mt) {
        #pragma unroll
        for (int nt = 0; nt < 2; ++nt) {
            const int N = n0 + wn * 32 + nt * 16 + ln;
            const float bn = bias[N];
            #pragma unroll
            for (int v = 0; v < 8; ++v) {
                const int Mi = m0 + wm * 64 + mt * 16 + v + 8 * hi;
                C[(size_t)Mi * D + N] = (bf16)(acc[mt][nt][v] + bn);
            }
        }
    }
}

// ---------------------------------------------------------------------------
// Kernel 2: flash attention, O accumulators in registers.
// One block per (batch, 64-query block); 512 threads = 16 waves.
// Wave (g, dch): g = query group (16 rows), dch = 128-wide d-chunk of output.
//   * scores: the 4 waves of group g split the d=512 reduction (4 K-steps each),
//     partial f32 accumulators reduced through LDS (contiguous b128 ops).
//   * softmax stats recomputed redundantly per wave (registers only).
//   * P -> bf16 staged in LDS (C-layout -> A-layout) by dch==0 wave.
//   * O update: register-resident v8f acc[8] per wave, rescaled by alpha, WMMA.
// LDS (~185 KB of 320 KB/WGP):
//   Qs [64][528] bf16 | Ks [32][528] bf16 | Vt [512][48] bf16
//   Pl 4x[16][48] bf16 | Sp 16x512 f32 (score partials)
// ---------------------------------------------------------------------------
static constexpr int QS_STRIDE = 528;
static constexpr int KS_STRIDE = 528;
static constexpr int VT_STRIDE = 48;
static constexpr int PL_STRIDE = 48;
static constexpr int PL_GROUP  = 16 * PL_STRIDE;

static constexpr size_t OFF_QS = 0;
static constexpr size_t OFF_KS = OFF_QS + (size_t)64 * QS_STRIDE * 2;   // 67584
static constexpr size_t OFF_VT = OFF_KS + (size_t)32 * KS_STRIDE * 2;   // +33792
static constexpr size_t OFF_PL = OFF_VT + (size_t)512 * VT_STRIDE * 2;  // +49152
static constexpr size_t OFF_SP = OFF_PL + (size_t)4 * PL_GROUP * 2;     // +6144
static constexpr size_t ATTN_SMEM = OFF_SP + (size_t)16 * 512 * 4;      // 189440 B

__device__ __forceinline__ void stage_b128(const bf16* g, bf16* l) {
#if USE_ASYNC_LDS
    __builtin_amdgcn_global_load_async_to_lds_b128(
        (gv4i_p)(bf16*)g,   // strip const, then generic->AS1 int4*
        (lv4i_p)l,          // generic->AS3 int4*
        0, 0);
#else
    *(uint4*)l = *(const uint4*)g;
#endif
}

__device__ __forceinline__ void stage_wait() {
#if USE_ASYNC_LDS
    __builtin_amdgcn_s_wait_asynccnt(0);
#endif
}

__global__ __launch_bounds__(512)
void flash_attn_bf16(const bf16* __restrict__ qp, const bf16* __restrict__ kp,
                     const bf16* __restrict__ vp, float* __restrict__ out) {
    extern __shared__ char smem[];
    bf16*  Qs = (bf16*)(smem + OFF_QS);
    bf16*  Ks = (bf16*)(smem + OFF_KS);
    bf16*  Vt = (bf16*)(smem + OFF_VT);
    bf16*  Pl = (bf16*)(smem + OFF_PL);
    float* Sp = (float*)(smem + OFF_SP);

    const int b    = blockIdx.y;
    const int q0   = blockIdx.x * 64;
    const int tid  = threadIdx.x;
    const int lane = tid & 31;
    const int wave = tid >> 5;       // 0..15
    const int ln   = lane & 15;
    const int hi   = lane >> 4;
    const int g    = wave >> 2;      // query group 0..3
    const int dch  = wave & 3;       // d-chunk 0..3 (128 wide)
    const int dw   = dch * 128;

    const float SCALE = 0.044194173824159216f;   // 1/sqrt(512)
    const float LOG2E = 1.4426950408889634f;

    // ---- preload Q block (64x512 bf16) via async-to-LDS ----
    const bf16* qbase = qp + ((size_t)b * S + q0) * D;
    for (int i = tid; i < 64 * 64; i += 512) {            // granule = 8 bf16
        const int r = i >> 6, c = i & 63;
        stage_b128(&qbase[(size_t)r * D + c * 8], &Qs[r * QS_STRIDE + c * 8]);
    }

    v8f acc_o[8];                    // O[16 rows x 128 d] for (g, dch), registers
    #pragma unroll
    for (int dc = 0; dc < 8; ++dc) acc_o[dc] = {};
    float m_run[8], l_run[8];
    #pragma unroll
    for (int v = 0; v < 8; ++v) { m_run[v] = -1.0e30f; l_run[v] = 0.0f; }

    for (int kt = 0; kt < S / 32; ++kt) {
        const int k0 = kt * 32;
        const bf16* kbase = kp + ((size_t)b * S + k0) * D;
        const bf16* vbase = vp + ((size_t)b * S + k0) * D;
        __syncthreads();   // previous round done with Ks/Vt/Pl

        // ---- stage K tile (async straight copy, B-layout as-is) ----
        for (int i = tid; i < 32 * 64; i += 512) {
            const int r = i >> 6, c = i & 63;
            stage_b128(&kbase[(size_t)r * D + c * 8], &Ks[r * KS_STRIDE + c * 8]);
        }
        // ---- stage V tile transposed: Vt[d][k] ----
        for (int i = tid; i < 32 * 64; i += 512) {
            const int r = i >> 6, c = i & 63;
            uint4 u = *(const uint4*)&vbase[(size_t)r * D + c * 8];
            const bf16* e = (const bf16*)&u;
            const int d0 = c * 8;
            #pragma unroll
            for (int j = 0; j < 8; ++j) Vt[(d0 + j) * VT_STRIDE + r] = e[j];
        }
        if (kt + 1 < S / 32) {       // L2 prefetch of next tiles
            __builtin_prefetch(kbase + (size_t)32 * D + (tid & 127) * 32, 0, 1);
            __builtin_prefetch(vbase + (size_t)32 * D + (tid & 127) * 32, 0, 1);
        }
        stage_wait();
        __syncthreads();

        // ---- partial scores: this wave covers d in [dch*128, dch*128+128) ----
        v8f s0 = {}, s1 = {};
        #pragma unroll
        for (int i = 0; i < 4; ++i) {
            const int dk = dch * 4 + i;
            v16bf a  = ldfrag(&Qs[(g * 16 + ln) * QS_STRIDE + dk * 32 + hi * 16]);
            v16bf b0 = ldfrag(&Ks[ln * KS_STRIDE + dk * 32 + hi * 16]);
            v16bf b1 = ldfrag(&Ks[(16 + ln) * KS_STRIDE + dk * 32 + hi * 16]);
            s0 = WMMA_BF16(a, b0, s0);
            s1 = WMMA_BF16(a, b1, s1);
        }
        {   // write partials (32B-contiguous per lane -> b128 DS stores)
            float* sp = &Sp[(size_t)wave * 512 + lane * 8];
            #pragma unroll
            for (int v = 0; v < 8; ++v) { sp[v] = s0[v]; sp[256 + v] = s1[v]; }
        }
        __syncthreads();

        // ---- reduce the 4 d-partials of my group (every wave, redundant) ----
        v8f t0 = {}, t1 = {};
        #pragma unroll
        for (int w2 = 0; w2 < 4; ++w2) {
            const float* sp = &Sp[(size_t)(g * 4 + w2) * 512 + lane * 8];
            #pragma unroll
            for (int v = 0; v < 8; ++v) { t0[v] += sp[v]; t1[v] += sp[256 + v]; }
        }

        // ---- online softmax (row = v + 8*hi spans one 16-lane half) ----
        float alpha_r[8];
        #pragma unroll
        for (int v = 0; v < 8; ++v) {
            const float x0 = t0[v] * SCALE;
            const float x1 = t1[v] * SCALE;
            float mx = fmaxf(x0, x1);
            #pragma unroll
            for (int msk = 1; msk <= 8; msk <<= 1) mx = fmaxf(mx, __shfl_xor(mx, msk, 32));
            const float mnew = fmaxf(m_run[v], mx);
            const float p0 = exp2f((x0 - mnew) * LOG2E);
            const float p1 = exp2f((x1 - mnew) * LOG2E);
            float sum = p0 + p1;
            #pragma unroll
            for (int msk = 1; msk <= 8; msk <<= 1) sum += __shfl_xor(sum, msk, 32);
            alpha_r[v] = exp2f((m_run[v] - mnew) * LOG2E);
            l_run[v] = l_run[v] * alpha_r[v] + sum;
            m_run[v] = mnew;
            if (dch == 0) {          // one wave per group stages P (A-layout rows)
                bf16* prow = &Pl[g * PL_GROUP + (v + 8 * hi) * PL_STRIDE];
                prow[ln]      = (bf16)p0;
                prow[16 + ln] = (bf16)p1;
            }
        }
        __syncthreads();

        // ---- O update: acc = acc*alpha + P(16x32) x V(32x128-chunk) ----
        v16bf pa = ldfrag(&Pl[g * PL_GROUP + ln * PL_STRIDE + hi * 16]);
        #pragma unroll
        for (int dc = 0; dc < 8; ++dc) {
            const int d0 = dw + dc * 16;
            v8f c;
            #pragma unroll
            for (int v = 0; v < 8; ++v) c[v] = acc_o[dc][v] * alpha_r[v];
            v16bf vb = ldfrag(&Vt[(d0 + ln) * VT_STRIDE + hi * 16]);
            acc_o[dc] = WMMA_BF16(pa, vb, c);
        }
    }

    // ---- normalize and store straight from registers ----
    float invl[8];
    #pragma unroll
    for (int v = 0; v < 8; ++v) invl[v] = 1.0f / l_run[v];
    float* obase = out + ((size_t)b * S + q0 + g * 16) * D + dw;
    #pragma unroll
    for (int dc = 0; dc < 8; ++dc) {
        #pragma unroll
        for (int v = 0; v < 8; ++v) {
            const int row = v + 8 * hi;
            obase[(size_t)row * D + dc * 16 + ln] = acc_o[dc][v] * invl[v];
        }
    }
}

// ---------------------------------------------------------------------------
// Host launcher
// ---------------------------------------------------------------------------
extern "C" void kernel_launch(void* const* d_in, const int* in_sizes, int n_in,
                              void* d_out, int out_size, void* d_ws, size_t ws_size,
                              hipStream_t stream) {
    (void)in_sizes; (void)n_in; (void)out_size; (void)ws_size;
    const float* q  = (const float*)d_in[0];
    const float* k  = (const float*)d_in[1];
    const float* v  = (const float*)d_in[2];
    const float* Wq = (const float*)d_in[3];
    const float* bq = (const float*)d_in[4];
    const float* Wk = (const float*)d_in[5];
    const float* bk = (const float*)d_in[6];
    const float* Wv = (const float*)d_in[7];
    const float* bv = (const float*)d_in[8];
    float* out = (float*)d_out;

    bf16* qp = (bf16*)d_ws;
    bf16* kp = qp + (size_t)M_TOT * D;
    bf16* vp = kp + (size_t)M_TOT * D;

    dim3 pg(D / 128, M_TOT / 128);   // 4 x 128
    proj_gemm_bf16<<<pg, 256, 0, stream>>>(q, Wq, bq, qp);
    proj_gemm_bf16<<<pg, 256, 0, stream>>>(k, Wk, bk, kp);
    proj_gemm_bf16<<<pg, 256, 0, stream>>>(v, Wv, bv, vp);

    dim3 ag(S / 64, NB);             // 32 x 8
    flash_attn_bf16<<<ag, 512, ATTN_SMEM, stream>>>(qp, kp, vp, out);
}
